// Spike_RNN_With_T_60902636257621
// MI455X (gfx1250) — matrix operand order repaired and checked by reading, not compiled, not run on hardware
//
#include <hip/hip_runtime.h>

// ---------------------------------------------------------------------------
// Spiking RNN (2 layers), CDNA5 / gfx1250, wave32 + WMMA bf16 + TDM.
//   L=256, T=4, B=32, C=H=512.  Rows per step R = T*B = 128.
// Pipeline:
//   1) pack W_* (f32 -> bf16, WMMA-B fragment layout)
//   2) pre0 = x @ Wih0 + b0          (WMMA GEMM, B staged in LDS via TDM)
//   3) scan layer0 (persistent, W_hh slice TDM-staged + LDS-resident)
//   4) pre1 = out0 @ Wih1 + b1
//   5) scan layer1 -> d_out (spikes f32, non-temporal) + new_state
// ---------------------------------------------------------------------------

#define USE_TDM 1

typedef __attribute__((ext_vector_type(16))) __bf16         v16bf;
typedef __attribute__((ext_vector_type(8)))  float          v8f;
typedef __attribute__((ext_vector_type(16))) unsigned short v16us;
typedef __attribute__((ext_vector_type(4)))  unsigned int   v4u;
typedef __attribute__((ext_vector_type(8)))  int            v8i;
typedef __attribute__((ext_vector_type(4)))  int            v4i;

#define L_   256
#define T_   4
#define B_   32
#define C_   512
#define H_   512
#define RW   128          // rows per timestep = T*B
#define LTB  32768        // L*T*B
#define NWG_SCAN 16

static __device__ __forceinline__ unsigned short f2bf(float f) {
  __bf16 h = (__bf16)f;                        // native cvt, RTE
  return __builtin_bit_cast(unsigned short, h);
}

// Load a 16-bit fragment whose two 8-element halves sit at p and p+stride.
static __device__ __forceinline__ v16bf load_bf_frag(const unsigned short* p, int stride) {
  v16us u;
  reinterpret_cast<uint4*>(&u)[0] = *reinterpret_cast<const uint4*>(p);
  reinterpret_cast<uint4*>(&u)[1] = *reinterpret_cast<const uint4*>(p + stride);
  return __builtin_bit_cast(v16bf, u);
}

// Same but from LDS (keeps addrspace(3) -> ds_load_b128).
static __device__ __forceinline__ v16bf load_bf_frag_lds(const unsigned short* p) {
  v16us u;
  reinterpret_cast<uint4*>(&u)[0] = *reinterpret_cast<const uint4*>(p);
  reinterpret_cast<uint4*>(&u)[1] = *reinterpret_cast<const uint4*>(p + 8);
  return __builtin_bit_cast(v16bf, u);
}

// Load A fragment from f32 row-major memory (convert to bf16 on the fly).
static __device__ __forceinline__ v16bf load_f32_frag(const float* p) {
  float4 q0 = reinterpret_cast<const float4*>(p)[0];
  float4 q1 = reinterpret_cast<const float4*>(p)[1];
  float4 q2 = *reinterpret_cast<const float4*>(p + 8);
  float4 q3 = *reinterpret_cast<const float4*>(p + 12);
  v16bf a;
  a[0]=(__bf16)q0.x; a[1]=(__bf16)q0.y; a[2]=(__bf16)q0.z; a[3]=(__bf16)q0.w;
  a[4]=(__bf16)q1.x; a[5]=(__bf16)q1.y; a[6]=(__bf16)q1.z; a[7]=(__bf16)q1.w;
  a[8]=(__bf16)q2.x; a[9]=(__bf16)q2.y; a[10]=(__bf16)q2.z; a[11]=(__bf16)q2.w;
  a[12]=(__bf16)q3.x; a[13]=(__bf16)q3.y; a[14]=(__bf16)q3.z; a[15]=(__bf16)q3.w;
  return a;
}

#if USE_TDM
// ---------------------------------------------------------------------------
// Tensor Data Mover: 2D tile load Global -> LDS.
// D# per CDNA5 ISA ch.8: group0 = {count/lds_addr/global_addr/type=2},
// group1 = {data_size=8B, tensor dims, tile dims, strides}, groups 2/3 = 0.
// All sizes below are in 8-byte units.
// ---------------------------------------------------------------------------
static __device__ __forceinline__ void tdm_load_2d(unsigned lds_off,
                                                   const void* gsrc,
                                                   unsigned tile_d0,
                                                   unsigned tile_d1,
                                                   unsigned stride0,
                                                   unsigned tensor_d0) {
  unsigned long long ga = (unsigned long long)gsrc;
  v4u g0;
  g0[0] = 1u;                                            // count=1, user mode
  g0[1] = lds_off;                                       // LDS byte address
  g0[2] = (unsigned)(ga & 0xFFFFFFFFull);                // global_addr[31:0]
  g0[3] = (unsigned)((ga >> 32) & 0x01FFFFFFull) | (2u << 30);  // [56:32] | type=2
  v8i g1;
  g1[0] = (int)(3u << 16);                               // data_size = 8B
  g1[1] = (int)((tensor_d0 & 0xFFFFu) << 16);            // tensor_dim0[15:0]
  g1[2] = (int)(((tensor_d0 >> 16) & 0xFFFFu) | ((tile_d1 & 0xFFFFu) << 16)); // dim0 hi | tensor_dim1 lo
  g1[3] = (int)((tile_d0 & 0xFFFFu) << 16);              // tensor_dim1 hi=0 | tile_dim0
  g1[4] = (int)(tile_d1 & 0xFFFFu);                      // tile_dim1 | tile_dim2=0
  g1[5] = (int)stride0;                                  // tensor_dim0_stride[31:0]
  g1[6] = 0;
  g1[7] = 0;
  v4i g2 = {0, 0, 0, 0};
  v4i g3 = {0, 0, 0, 0};
#if defined(__clang_major__) && (__clang_major__ >= 23)
  v8i g4 = {0, 0, 0, 0, 0, 0, 0, 0};
  __builtin_amdgcn_tensor_load_to_lds(g0, g1, g2, g3, g4, 0);
#else
  __builtin_amdgcn_tensor_load_to_lds(g0, g1, g2, g3, 0);
#endif
}
#endif  // USE_TDM

// ---------------------------------------------------------------------------
// Pack a [K x N] row-major f32 weight into WMMA-B bf16 blocks.
// Block (kt, nt) = 1KB: lane l holds B[kt*32 + (l>>4)*16 + e, nt*16 + (l&15)],
// e = 0..15, contiguous per lane (32 B).
// ---------------------------------------------------------------------------
__global__ void pack_b_kernel(const float* __restrict__ W,
                              unsigned short* __restrict__ out, int K, int N) {
  int idx = blockIdx.x * blockDim.x + threadIdx.x;
  if (idx >= K * N) return;
  int blk  = idx >> 9;           // 512 elems per block
  int rem  = idx & 511;
  int lane = rem >> 4;
  int e    = rem & 15;
  int nTiles = N >> 4;
  int kt = blk / nTiles, nt = blk - kt * nTiles;
  int k = kt * 32 + ((lane >> 4) << 4) + e;
  int n = nt * 16 + (lane & 15);
  out[idx] = f2bf(W[(size_t)k * N + n]);
}

// Convert initial hidden state slabs to bf16, zero barrier counters.
__global__ void init_kernel(const float* __restrict__ st,
                            unsigned short* __restrict__ h0,
                            unsigned short* __restrict__ h1,
                            unsigned* __restrict__ bars) {
  int idx = blockIdx.x * blockDim.x + threadIdx.x;
  if (idx < RW * H_) {
    h0[idx] = f2bf(st[idx]);
    h1[idx] = f2bf(st[RW * H_ + idx]);
  }
  if (idx < 8) bars[idx] = 0u;
}

// ---------------------------------------------------------------------------
// GEMM: out[M=LTB, H] = A[M, 512] @ Bp(packed) + bias, stored in C-fragment
// layout: out[(mt*(H/16)+nt)*256 + lane*8 + r].
// Grid: 2048 blocks x 256 threads. WG b: nq = b&7 (4 nTiles, B slice staged
// into 64KB LDS via TDM); its 8 waves cover mt = (b>>3)*8 + wave.
// ---------------------------------------------------------------------------
template <bool A_IS_F32>
__global__ void gemm_bias_kernel(const void* __restrict__ Aptr,
                                 const unsigned short* __restrict__ Bp,
                                 const float* __restrict__ bias,
                                 float* __restrict__ outp) {
  __shared__ __align__(16) unsigned short ldsB[64 * 512];   // 64 KB
  const int tid  = threadIdx.x;
  const int lane = tid & 31;
  const int nq   = blockIdx.x & 7;
  const int mt   = (blockIdx.x >> 3) * 8 + (tid >> 5);
  const int half = lane >> 4;
  const int ml   = lane & 15;
  const int row  = mt * 16 + ml;

  // Stage this WG's B slice: 16 kt rows x 4KB contiguous, 32KB row stride.
#if USE_TDM
  if (tid == 0) {
    tdm_load_2d((unsigned)(uintptr_t)&ldsB[0],
                Bp + ((size_t)(nq * 4) << 9),
                /*tile_d0=*/512, /*tile_d1=*/16,
                /*stride0=*/4096, /*tensor_d0=*/4096);
    __builtin_amdgcn_s_wait_tensorcnt(0);
  }
  __syncthreads();
#else
  for (int c = tid; c < 4096; c += 256) {
    const int kt  = c >> 8;
    const int rem = c & 255;
    const unsigned short* src = Bp + ((size_t)(kt * 32 + nq * 4) << 9) + rem * 8;
    *reinterpret_cast<uint4*>(&ldsB[(size_t)kt * 2048 + rem * 8]) =
        *reinterpret_cast<const uint4*>(src);
  }
  __syncthreads();
#endif

  v8f c0 = {}, c1 = {}, c2 = {}, c3 = {};
#pragma unroll 4
  for (int kt = 0; kt < C_ / 32; ++kt) {
    const int k0 = kt * 32;
    v16bf a;
    if (A_IS_F32) {
      a = load_f32_frag((const float*)Aptr + (size_t)row * C_ + k0 + half * 16);
    } else {
      a = load_bf_frag((const unsigned short*)Aptr + (size_t)row * C_ + k0 + half * 16, 8);
    }
    const unsigned short* lb = &ldsB[(size_t)kt * 2048 + lane * 16];
    v16bf b0v = load_bf_frag_lds(lb);
    v16bf b1v = load_bf_frag_lds(lb + 512);
    v16bf b2v = load_bf_frag_lds(lb + 1024);
    v16bf b3v = load_bf_frag_lds(lb + 1536);
    c0 = __builtin_amdgcn_wmma_f32_16x16x32_bf16(false, a, false, b0v, (short)0, c0, false, false);
    c1 = __builtin_amdgcn_wmma_f32_16x16x32_bf16(false, a, false, b1v, (short)0, c1, false, false);
    c2 = __builtin_amdgcn_wmma_f32_16x16x32_bf16(false, a, false, b2v, (short)0, c2, false, false);
    c3 = __builtin_amdgcn_wmma_f32_16x16x32_bf16(false, a, false, b3v, (short)0, c3, false, false);
  }
  v8f acc[4] = {c0, c1, c2, c3};
#pragma unroll
  for (int j = 0; j < 4; ++j) {
    const int nt = nq * 4 + j;
    const float bb = bias[nt * 16 + ml];
    float* op = outp + (((size_t)mt * (H_ >> 4) + nt) << 8) + lane * 8;
    float4 s0 = make_float4(acc[j][0] + bb, acc[j][1] + bb, acc[j][2] + bb, acc[j][3] + bb);
    float4 s1 = make_float4(acc[j][4] + bb, acc[j][5] + bb, acc[j][6] + bb, acc[j][7] + bb);
    reinterpret_cast<float4*>(op)[0] = s0;
    reinterpret_cast<float4*>(op)[1] = s1;
  }
}

// Global arrive-and-spin barrier (monotonic counter; one counter per scan).
static __device__ __forceinline__ void grid_sync(unsigned* bar, int step, unsigned nwg) {
  __threadfence();
  __syncthreads();
  if (threadIdx.x == 0) {
    __hip_atomic_fetch_add(bar, 1u, __ATOMIC_ACQ_REL, __HIP_MEMORY_SCOPE_AGENT);
    const unsigned target = nwg * (unsigned)(step + 1);
    while (__hip_atomic_load(bar, __ATOMIC_ACQUIRE, __HIP_MEMORY_SCOPE_AGENT) < target) {
      __builtin_amdgcn_s_sleep(2);
    }
  }
  __syncthreads();
  __threadfence();
}

// ---------------------------------------------------------------------------
// Persistent recurrent scan: for l in 0..L-1: h = spike(pre[l] + h_prev@Whh).
// Grid: NWG_SCAN=16 blocks x 256 threads. WG b owns nTile pair np=b; its W_hh
// slice (2 nTiles x 16 kTiles = 32KB) is TDM-staged and LDS-resident for the
// whole scan. Wave w of the WG owns mTile mt=w.
// ---------------------------------------------------------------------------
__global__ void scan_kernel(const float* __restrict__ prep,          // [L][256 blk][32][8]
                            const unsigned short* __restrict__ Wp,   // packed Whh
                            const unsigned short* __restrict__ hinit,// [RW][H] bf16
                            unsigned short* __restrict__ hbuf,
                            int dbuf,
                            float* __restrict__ out32,               // nullable [L][RW][H]
                            float* __restrict__ state_out,           // [RW][H]
                            unsigned* __restrict__ bar) {
  __shared__ __align__(16) unsigned short ldsW[32 * 512];   // 32 KB
  const int tid  = threadIdx.x;
  const int lane = tid & 31;
  const int mt   = tid >> 5;        // 0..7
  const int np   = blockIdx.x;      // 0..15
  const int half = lane >> 4;
  const int ml   = lane & 15;
  const int row  = mt * 16 + ml;
  const size_t SLAB = (size_t)RW * H_;

  // One-time stage of this WG's W_hh slice: blocks (kt, np*2 + j) ->
  // 16 rows x 2KB contiguous, 32KB row stride.
#if USE_TDM
  if (tid == 0) {
    tdm_load_2d((unsigned)(uintptr_t)&ldsW[0],
                Wp + ((size_t)(np * 2) << 9),
                /*tile_d0=*/256, /*tile_d1=*/16,
                /*stride0=*/4096, /*tensor_d0=*/4096);
    __builtin_amdgcn_s_wait_tensorcnt(0);
  }
  __syncthreads();
#else
  for (int c = tid; c < 2048; c += 256) {
    const int ktj = c >> 6;               // kt*2 + j
    const int kt  = ktj >> 1, j = ktj & 1;
    const int off = (c & 63) * 8;
    const unsigned short* src = Wp + ((size_t)(kt * 32 + np * 2 + j) << 9) + off;
    *reinterpret_cast<uint4*>(&ldsW[(size_t)ktj * 512 + off]) =
        *reinterpret_cast<const uint4*>(src);
  }
  __syncthreads();
#endif

  for (int l = 0; l < L_; ++l) {
    const unsigned short* hsrc =
        (l == 0) ? hinit
                 : hbuf + (dbuf ? (size_t)((l - 1) & 1) * SLAB : (size_t)(l - 1) * SLAB);
    v8f c0 = {}, c1 = {};
#pragma unroll 4
    for (int kt = 0; kt < H_ / 32; ++kt) {
      v16bf a = load_bf_frag(hsrc + (size_t)row * H_ + kt * 32 + half * 16, 8);
      const unsigned short* lw = &ldsW[(size_t)(kt * 2) * 512 + lane * 16];
      v16bf b0v = load_bf_frag_lds(lw);
      v16bf b1v = load_bf_frag_lds(lw + 512);
      c0 = __builtin_amdgcn_wmma_f32_16x16x32_bf16(false, a, false, b0v, (short)0, c0, false, false);
      c1 = __builtin_amdgcn_wmma_f32_16x16x32_bf16(false, a, false, b1v, (short)0, c1, false, false);
    }
    unsigned short* hdst = hbuf + (dbuf ? (size_t)(l & 1) * SLAB : (size_t)l * SLAB);
    v8f acc[2] = {c0, c1};
#pragma unroll
    for (int j = 0; j < 2; ++j) {
      const int nt  = np * 2 + j;
      const int col = nt * 16 + ml;
      const float* pp = prep + (size_t)l * SLAB + (((size_t)mt * (H_ >> 4) + nt) << 8) + lane * 8;
      float4 p0 = reinterpret_cast<const float4*>(pp)[0];
      float4 p1 = reinterpret_cast<const float4*>(pp)[1];
      float pv[8] = {p0.x, p0.y, p0.z, p0.w, p1.x, p1.y, p1.z, p1.w};
#pragma unroll
      for (int r = 0; r < 8; ++r) {
        const float v = pv[r] + acc[j][r];
        const float s = (v >= 0.0f) ? 1.0f : 0.0f;
        const int orow = mt * 16 + half * 8 + r;   // C/D layout: M = r + half*8
        hdst[(size_t)orow * H_ + col] = (v >= 0.0f) ? (unsigned short)0x3F80 : (unsigned short)0;
        if (out32)
          __builtin_nontemporal_store(s, &out32[(size_t)l * SLAB + (size_t)orow * H_ + col]);
        if (l == L_ - 1)
          __builtin_nontemporal_store(s, &state_out[(size_t)orow * H_ + col]);
      }
      // Prefetch next step's pre-activation fragment across the barrier.
      if (l + 1 < L_) {
        __builtin_prefetch(prep + (size_t)(l + 1) * SLAB +
                               (((size_t)mt * (H_ >> 4) + nt) << 8) + lane * 8, 0, 0);
      }
    }
    grid_sync(bar, l, NWG_SCAN);
  }
}

// ---------------------------------------------------------------------------
extern "C" void kernel_launch(void* const* d_in, const int* in_sizes, int n_in,
                              void* d_out, int out_size, void* d_ws, size_t ws_size,
                              hipStream_t stream) {
  (void)in_sizes; (void)n_in; (void)out_size; (void)ws_size;
  const float* x     = (const float*)d_in[0];
  const float* state = (const float*)d_in[1];
  const float* Wih0  = (const float*)d_in[2];
  const float* Whh0  = (const float*)d_in[3];
  const float* b0    = (const float*)d_in[4];
  const float* Wih1  = (const float*)d_in[5];
  const float* Whh1  = (const float*)d_in[6];
  const float* b1    = (const float*)d_in[7];

  float* out1   = (float*)d_out;                       // [L,T,B,H]
  float* state0 = out1 + (size_t)LTB * H_;             // [T,B,H]
  float* state1 = state0 + (size_t)RW * H_;            // [T,B,H]

  char* w = (char*)d_ws;
  auto alloc = [&](size_t bytes) {
    char* p = w;
    w += (bytes + 255) & ~(size_t)255;
    return p;
  };
  unsigned short* wih0p  = (unsigned short*)alloc((size_t)C_ * H_ * 2);
  unsigned short* whh0p  = (unsigned short*)alloc((size_t)H_ * H_ * 2);
  unsigned short* wih1p  = (unsigned short*)alloc((size_t)H_ * H_ * 2);
  unsigned short* whh1p  = (unsigned short*)alloc((size_t)H_ * H_ * 2);
  unsigned short* hinit0 = (unsigned short*)alloc((size_t)RW * H_ * 2);
  unsigned short* hinit1 = (unsigned short*)alloc((size_t)RW * H_ * 2);
  unsigned short* h1buf  = (unsigned short*)alloc((size_t)2 * RW * H_ * 2);
  unsigned*       bars   = (unsigned*)alloc(256);
  float*          pre0p  = (float*)alloc((size_t)LTB * H_ * 4);   // 64 MB
  float*          pre1p  = (float*)alloc((size_t)LTB * H_ * 4);   // 64 MB
  unsigned short* out0b  = (unsigned short*)alloc((size_t)LTB * H_ * 2); // 32 MB

  const int PACK_BLOCKS = (C_ * H_ + 255) / 256;
  pack_b_kernel<<<PACK_BLOCKS, 256, 0, stream>>>(Wih0, wih0p, C_, H_);
  pack_b_kernel<<<PACK_BLOCKS, 256, 0, stream>>>(Whh0, whh0p, H_, H_);
  pack_b_kernel<<<PACK_BLOCKS, 256, 0, stream>>>(Wih1, wih1p, H_, H_);
  pack_b_kernel<<<PACK_BLOCKS, 256, 0, stream>>>(Whh1, whh1p, H_, H_);
  init_kernel<<<(RW * H_ + 255) / 256, 256, 0, stream>>>(state, hinit0, hinit1, bars);

  // Layer 0: pre0 = x @ Wih0 + b0, then recurrent scan.
  gemm_bias_kernel<true><<<LTB / 16, 256, 0, stream>>>(x, wih0p, b0, pre0p);
  scan_kernel<<<NWG_SCAN, 256, 0, stream>>>(pre0p, whh0p, hinit0, out0b,
                                            /*dbuf=*/0, /*out32=*/nullptr, state0, bars + 0);

  // Layer 1: pre1 = out0 @ Wih1 + b1, then recurrent scan -> d_out.
  gemm_bias_kernel<false><<<LTB / 16, 256, 0, stream>>>(out0b, wih1p, b1, pre1p);
  scan_kernel<<<NWG_SCAN, 256, 0, stream>>>(pre1p, whh1p, hinit1, h1buf,
                                            /*dbuf=*/1, /*out32=*/out1, state1, bars + 4);
}
